// BiGRU_80272938762324
// MI455X (gfx1250) — matrix operand order repaired and checked by reading, not compiled, not run on hardware
//
#include <hip/hip_runtime.h>

typedef _Float16 half_t;
typedef __attribute__((ext_vector_type(16))) _Float16 v16h;
typedef __attribute__((ext_vector_type(8)))  _Float16 v8h;
typedef __attribute__((ext_vector_type(8)))  float    v8f;

#define SEQ_T 8192
#define SEQ_B 32
#define TBROWS (SEQ_T * SEQ_B)   // 262144 flattened rows (t*B + b)

__device__ __forceinline__ float fast_rcp(float x) { return __builtin_amdgcn_rcpf(x); }
__device__ __forceinline__ float fast_sigmoid(float v) {
    return fast_rcp(1.0f + __expf(-v));             // v_exp + v_add + v_rcp
}
__device__ __forceinline__ float fast_tanh(float v) {
    float e = __expf(-2.0f * fabsf(v));             // sign-safe tanh via exp
    float r = (1.0f - e) * fast_rcp(1.0f + e);
    return copysignf(r, v);
}

// ---------------------------------------------------------------------------
// Input projection. Computes in[row] @ wi^T + bi (+bh for r,z gates) and
// stores f16 in *scan-lane order*: for step t, scan-thread st of batch-half mt
// reads its gate inputs (gate[3] x r[8]) as 3 contiguous 16B chunks.
//   xp[ (t*32W + st)*48 + mt*24 + gt*8 + r ]
// One wave per 16-row M tile; Din<=32 -> single K-step of 16x16x32 f16 WMMA.
// ---------------------------------------------------------------------------
template<int DIN, int H>
__global__ __launch_bounds__(256) void proj_kernel(
        const float* __restrict__ in,       // (TBROWS, DIN)
        const float* __restrict__ wi_f, const float* __restrict__ bi_f, const float* __restrict__ bh_f,
        const float* __restrict__ wi_b, const float* __restrict__ bi_b, const float* __restrict__ bh_b,
        half_t* __restrict__ xp_f, half_t* __restrict__ xp_b)
{
    constexpr int THREEH = 3 * H;
    constexpr int NT = THREEH / 16;
    constexpr int W  = H / 16;              // scan waves / column sub-blocks
    const int dir = blockIdx.y;
    const float* __restrict__ wi = dir ? wi_b : wi_f;
    const float* __restrict__ bi = dir ? bi_b : bi_f;
    const float* __restrict__ bh = dir ? bh_b : bh_f;
    half_t* __restrict__ xp = dir ? xp_b : xp_f;

    const int wave = threadIdx.x >> 5;
    const int lane = threadIdx.x & 31;
    const int g    = lane >> 4;
    const int ln   = lane & 15;
    const int row0 = (blockIdx.x * 8 + wave) * 16;
    const int t    = row0 >> 5;             // 16-row tile = one batch half of one t
    const int mt   = (row0 >> 4) & 1;       // which batch half (b 0-15 / 16-31)

    // Stage 16 x DIN A tile (zero-padded to 32 cols) in LDS, coalesced.
    __shared__ float atile[8][16 * 32];
    float* at = atile[wave];
    for (int i = lane; i < 16 * 32; i += 32) {
        const int r = i >> 5, c = i & 31;
        at[i] = (c < DIN) ? in[(long)(row0 + r) * DIN + c] : 0.0f;
    }
    __syncthreads();

    // A fragment: 16-bit A 16x32 layout (ISA 7.12.2).
    v16h a;
    #pragma unroll
    for (int e = 0; e < 16; ++e) {
        const int v = e >> 1, p = e & 1;
        const int k = ((v < 4) ? 0 : 16) + g * 8 + 2 * (v & 3) + p;
        a[e] = (half_t)at[ln * 32 + k];
    }

    #pragma unroll
    for (int nt = 0; nt < NT; ++nt) {
        const int col = nt * 16 + ln;       // column in [0, 3H)
        v16h bf;                            // B[k][col] = wi[col][k]
        #pragma unroll
        for (int e = 0; e < 16; ++e) {
            const int K = ((e < 8) ? 0 : 16) + g * 8 + (e & 7);
            bf[e] = (K < DIN) ? (half_t)wi[col * DIN + K] : (half_t)0.0f;
        }
        v8f c = {};
        c = __builtin_amdgcn_wmma_f32_16x16x32_f16(false, a, false, bf,
                                                   (short)0, c, false, false);
        // bias: bi always; bh folded in for r,z gates (cols < 2H)
        const float bv = bi[col] + ((col < 2 * H) ? bh[col] : 0.0f);
        // scan-lane layout store: one packed 16B store per tile
        const int gt = nt / W, w = nt % W;
        const int st = w * 32 + ln + 16 * g;     // scan thread owning this column
        v8h hv;
        #pragma unroll
        for (int r = 0; r < 8; ++r) hv[r] = (half_t)(c[r] + bv);
        *(v8h*)(xp + ((long)t * (32 * W) + st) * 48 + mt * 24 + gt * 8) = hv;
    }
}

// ---------------------------------------------------------------------------
// Recurrent scan. Grid = (2 dirs) x (2 batch halves); the recurrence is
// batch-independent, so each 16-row half runs in its own block (M=16 is the
// WMMA floor). WAVES = H/16 waves split the hidden columns; wave w owns
// N-tiles {gt*W + w}, so every lane holds the r/z/n partials of its 8 (b, j)
// elements in registers -> gates fully in-register, no hp LDS.
// Only LDS per step: h stored as f16 directly in A-fragment layout.
// xp loads are software-pipelined one step ahead.
// ---------------------------------------------------------------------------
template<int H, int WAVES>   // WAVES == H/16
__global__ __launch_bounds__(WAVES * 32) void scan_kernel(
        const half_t* __restrict__ xp_f, const half_t* __restrict__ xp_b,
        const float* __restrict__ wh_f, const float* __restrict__ bh_f,
        const float* __restrict__ wh_b, const float* __restrict__ bh_b,
        float* __restrict__ y)              // (TBROWS, 2H)
{
    constexpr int KS = (H + 31) / 32;       // 1 for H=16, 2 for H=64
    const int dir = blockIdx.x;             // 0 = fwd, 1 = bwd
    const int bh2 = blockIdx.y;             // batch half: rows b in [bh2*16, +16)
    const half_t* __restrict__ xp = dir ? xp_b : xp_f;
    const float*  __restrict__ wh = dir ? wh_b : wh_f;
    const float*  __restrict__ bh = dir ? bh_b : bh_f;

    const int tid  = threadIdx.x;
    const int w    = tid >> 5;
    const int lane = tid & 31;
    const int g    = lane >> 4;             // half-wave group
    const int ln   = lane & 15;
    const int j    = w * 16 + ln;           // this lane's hidden column

    __shared__ v16h h16[KS * 32];           // h (16 batches) in f16 A-frag layout

    {   // zero init (covers K-padding for H=16 permanently)
        half_t* hz = (half_t*)h16;
        for (int i = tid; i < KS * 32 * 16; i += WAVES * 32) hz[i] = (half_t)0.0f;
    }

    // Wh^T B-fragments for this wave's tiles (nt_global = gt*W + w), f16, once.
    v16h bfr[3][KS];
    #pragma unroll
    for (int gt = 0; gt < 3; ++gt) {
        const int n = gt * H + j;           // column in [0, 3H)
        #pragma unroll
        for (int ks = 0; ks < KS; ++ks) {
            #pragma unroll
            for (int e = 0; e < 16; ++e) {
                const int K = ks * 32 + ((e < 8) ? 0 : 16) + g * 8 + (e & 7);
                bfr[gt][ks][e] = (K < H) ? (half_t)wh[n * H + K] : (half_t)0.0f;
            }
        }
    }
    const float bhn = bh[2 * H + j];        // only n-gate bias survives folding

    // h16 write position for this lane (constant over t): k = j
    const int ksj = j >> 5;
    const int ew  = (((j & 31) >= 16) ? 8 : 0) + (ln & 7);
    const int gw  = ln >> 3;

    float hreg[8];                          // f32 carry, in registers
    #pragma unroll
    for (int r = 0; r < 8; ++r) hreg[r] = 0.0f;

    // prefetch xp for first step (this batch half's 3 gate chunks)
    v8h px[3];                              // [gt]
    {
        const long tt0 = dir ? (SEQ_T - 1) : 0;
        const v8h* p = (const v8h*)(xp + (tt0 * (32 * WAVES) + tid) * 48 + bh2 * 24);
        #pragma unroll
        for (int c2 = 0; c2 < 3; ++c2) px[c2] = p[c2];
    }
    if (WAVES > 1) __syncthreads();

    for (int t = 0; t < SEQ_T; ++t) {
        const long ttc = dir ? (SEQ_T - 1 - t) : t;

        if (WAVES > 1) __syncthreads();     // h16 stores from t-1 visible
        v16h afr[KS];
        #pragma unroll
        for (int ks = 0; ks < KS; ++ks)
            afr[ks] = h16[ks * 32 + lane];
        if (WAVES > 1) __syncthreads();     // all reads done; safe to overwrite

        v8f acc[3];
        #pragma unroll
        for (int gt = 0; gt < 3; ++gt) {
            v8f c = {};
            #pragma unroll
            for (int ks = 0; ks < KS; ++ks)
                c = __builtin_amdgcn_wmma_f32_16x16x32_f16(false, afr[ks],
                        false, bfr[gt][ks], (short)0, c, false, false);
            acc[gt] = c;
        }

        // gates fully in registers; lane owns (b_local = r + 8g, j)
        half_t* hw = (half_t*)h16;
        const int slot0 = ksj * 32 + gw * 16;
        float* yb = y + (ttc * SEQ_B + bh2 * 16 + 8 * g) * (2 * H) + (long)dir * H + j;
        #pragma unroll
        for (int r = 0; r < 8; ++r) {
            const float xr = (float)px[0][r];
            const float xz = (float)px[1][r];
            const float xn = (float)px[2][r];
            const float rg = fast_sigmoid(xr + acc[0][r]);
            const float zg = fast_sigmoid(xz + acc[1][r]);
            const float ng = fast_tanh(xn + rg * (acc[2][r] + bhn));
            const float hnew = (1.0f - zg) * ng + zg * hreg[r];
            hreg[r] = hnew;
            hw[(slot0 + r + 8 * g) * 16 + ew] = (half_t)hnew;
            yb[(long)r * (2 * H)] = hnew;
        }

        // software-pipelined xp prefetch for next step
        const long ttn = dir ? ((t + 1 < SEQ_T) ? (SEQ_T - 2 - t) : 0)
                             : ((t + 1 < SEQ_T) ? (t + 1) : (SEQ_T - 1));
        const v8h* p = (const v8h*)(xp + (ttn * (32 * WAVES) + tid) * 48 + bh2 * 24);
        #pragma unroll
        for (int c2 = 0; c2 < 3; ++c2) px[c2] = p[c2];
    }
}

// ---------------------------------------------------------------------------
// dense+relu: (TBROWS,32) @ (8,32)^T + b, relu. One thread per row.
// ---------------------------------------------------------------------------
__global__ __launch_bounds__(256) void dense_relu_kernel(
        const float* __restrict__ in, const float* __restrict__ w,
        const float* __restrict__ b, float* __restrict__ out)
{
    const long row = (long)blockIdx.x * blockDim.x + threadIdx.x;
    const float4* inr = (const float4*)(in + row * 32);
    float4 xv[8];
    #pragma unroll
    for (int q = 0; q < 8; ++q) xv[q] = inr[q];
    #pragma unroll
    for (int o = 0; o < 8; ++o) {
        const float4* wr = (const float4*)(w + o * 32);
        float acc = b[o];
        #pragma unroll
        for (int q = 0; q < 8; ++q) {
            const float4 wv = wr[q];
            acc += xv[q].x * wv.x + xv[q].y * wv.y + xv[q].z * wv.z + xv[q].w * wv.w;
        }
        out[row * 8 + o] = fmaxf(acc, 0.0f);
    }
}

// ---------------------------------------------------------------------------
// Final projection: (TBROWS,128) @ (1,128)^T + b. One wave per row.
// ---------------------------------------------------------------------------
__global__ __launch_bounds__(256) void outproj_kernel(
        const float* __restrict__ in, const float* __restrict__ w,
        const float* __restrict__ b, float* __restrict__ out)
{
    const int  wave = threadIdx.x >> 5;
    const int  lane = threadIdx.x & 31;
    const long row  = (long)blockIdx.x * 8 + wave;
    const float4 a  = *(const float4*)(in + row * 128 + lane * 4);
    const float4 ww = *(const float4*)(w + lane * 4);
    float acc = a.x * ww.x + a.y * ww.y + a.z * ww.z + a.w * ww.w;
    #pragma unroll
    for (int off = 16; off > 0; off >>= 1)
        acc += __shfl_down(acc, off, 32);
    if (lane == 0) out[row] = acc + b[0];
}

// ---------------------------------------------------------------------------
extern "C" void kernel_launch(void* const* d_in, const int* in_sizes, int n_in,
                              void* d_out, int out_size, void* d_ws, size_t ws_size,
                              hipStream_t stream) {
    (void)in_sizes; (void)n_in; (void)out_size; (void)ws_size;
    const float* x = (const float*)d_in[0];
    // per-GRU param base + {0:wif,1:whf,2:bif,3:bhf,4:wib,5:whb,6:bib,7:bhb}
    const int G1 = 1, G2 = 9, G3 = 17, G4 = 25;
    auto P = [&](int i) { return (const float*)d_in[i]; };
    const float* dense_w = P(33); const float* dense_b = P(34);
    const float* out_w   = P(35); const float* out_b   = P(36);

    // workspace layout (~444 MB)
    float*  y1  = (float*)d_ws;
    float*  y2  = y1 + (size_t)TBROWS * 32;
    float*  yd  = y2 + (size_t)TBROWS * 32;
    float*  y3  = yd + (size_t)TBROWS * 8;
    float*  y4  = y3 + (size_t)TBROWS * 32;
    half_t* xpf = (half_t*)(y4 + (size_t)TBROWS * 128);
    half_t* xpb = xpf + (size_t)TBROWS * 192;   // max layout: H=64 -> T*128*48 halves

    const dim3 pgrid(TBROWS / 128, 2);   // 8 waves x 16 rows per block, 2 dirs
    const dim3 sgrid(2, 2);              // dirs x batch halves

    // g1: Din=32, H=16
    proj_kernel<32, 16><<<pgrid, 256, 0, stream>>>(x,  P(G1+0), P(G1+2), P(G1+3),
                                                       P(G1+4), P(G1+6), P(G1+7), xpf, xpb);
    scan_kernel<16, 1><<<sgrid, 32, 0, stream>>>(xpf, xpb, P(G1+1), P(G1+3), P(G1+5), P(G1+7), y1);

    // g2: Din=32, H=16
    proj_kernel<32, 16><<<pgrid, 256, 0, stream>>>(y1, P(G2+0), P(G2+2), P(G2+3),
                                                       P(G2+4), P(G2+6), P(G2+7), xpf, xpb);
    scan_kernel<16, 1><<<sgrid, 32, 0, stream>>>(xpf, xpb, P(G2+1), P(G2+3), P(G2+5), P(G2+7), y2);

    // dense + relu
    dense_relu_kernel<<<TBROWS / 256, 256, 0, stream>>>(y2, dense_w, dense_b, yd);

    // g3: Din=8, H=16
    proj_kernel<8, 16><<<pgrid, 256, 0, stream>>>(yd,  P(G3+0), P(G3+2), P(G3+3),
                                                       P(G3+4), P(G3+6), P(G3+7), xpf, xpb);
    scan_kernel<16, 1><<<sgrid, 32, 0, stream>>>(xpf, xpb, P(G3+1), P(G3+3), P(G3+5), P(G3+7), y3);

    // g4: Din=32, H=64
    proj_kernel<32, 64><<<pgrid, 256, 0, stream>>>(y3, P(G4+0), P(G4+2), P(G4+3),
                                                       P(G4+4), P(G4+6), P(G4+7), xpf, xpb);
    scan_kernel<64, 4><<<sgrid, 128, 0, stream>>>(xpf, xpb, P(G4+1), P(G4+3), P(G4+5), P(G4+7), y4);

    // final 128 -> 1 projection
    outproj_kernel<<<TBROWS / 8, 256, 0, stream>>>(y4, out_w, out_b, (float*)d_out);
}
